// PointNet2Comp_17291538333861
// MI455X (gfx1250) — matrix operand order repaired and checked by reading, not compiled, not run on hardware
//
#include <hip/hip_runtime.h>

typedef _Float16 half_t;
typedef __attribute__((ext_vector_type(16))) _Float16 v16h;
typedef __attribute__((ext_vector_type(8)))  _Float16 v8h;
typedef __attribute__((ext_vector_type(8)))  float    v8f;

#define BN_EPS 1e-5f
#define MODE_STORE 0
#define MODE_POOL  1

// ---------------------------------------------------------------------------
// Weight convert f32 -> f16, zero-padded to Kp columns (Kp multiple of 32).
// ---------------------------------------------------------------------------
__global__ void k_cvtw(const float* __restrict__ W, half_t* __restrict__ Wh,
                       int M, int K, int Kp) {
  int idx = blockIdx.x * 256 + threadIdx.x;
  if (idx >= M * Kp) return;
  int m = idx / Kp, k = idx % Kp;
  Wh[idx] = (k < K) ? (half_t)W[(size_t)m * K + k] : (half_t)0.0f;
}

// ---------------------------------------------------------------------------
// Farthest point sampling: one 256-thread block per batch. Entire point cloud
// (px/py/pz) plus the running min-distance array live in LDS (dynamic smem,
// ~66KB for N=4096; MI455X has 320KB LDS per WGP). 512 serial argmax steps
// then never touch global memory except to record results.
// ---------------------------------------------------------------------------
__global__ __launch_bounds__(256) void k_fps(const float* __restrict__ xyz, int N, int P,
                                             int* __restrict__ fidx, float* __restrict__ nxyz) {
  extern __shared__ float smem[];
  float* dist = smem;            // N
  float* px   = dist + N;        // N
  float* py   = px + N;          // N
  float* pz   = py + N;          // N
  float* rv   = pz + N;          // 256
  int*   ri   = (int*)(rv + 256);// 256
  float* cp   = (float*)(ri + 256); // 3

  int b = blockIdx.x, tid = threadIdx.x;
  const float* pb = xyz + (size_t)b * N * 3;
  float p0x = pb[0], p0y = pb[1], p0z = pb[2];
  for (int j = tid; j < N; j += 256) {
    float x_ = pb[j*3+0], y_ = pb[j*3+1], z_ = pb[j*3+2];
    px[j] = x_; py[j] = y_; pz[j] = z_;
    float dx = x_-p0x, dy = y_-p0y, dz = z_-p0z;
    dist[j] = dx*dx + dy*dy + dz*dz;
  }
  if (tid == 0) {
    fidx[(size_t)b*P] = 0;
    nxyz[((size_t)b*P)*3+0] = p0x;
    nxyz[((size_t)b*P)*3+1] = p0y;
    nxyz[((size_t)b*P)*3+2] = p0z;
  }
  __syncthreads();
  for (int it = 1; it < P; ++it) {
    float bv = -1.0f; int bi = 0;
    for (int j = tid; j < N; j += 256) {
      float d = dist[j];
      if (d > bv) { bv = d; bi = j; }   // strict >: earliest index kept
    }
    rv[tid] = bv; ri[tid] = bi;
    __syncthreads();
    for (int off = 128; off > 0; off >>= 1) {
      if (tid < off) {
        float ov = rv[tid+off]; int oi = ri[tid+off];
        if (ov > rv[tid] || (ov == rv[tid] && oi < ri[tid])) { rv[tid] = ov; ri[tid] = oi; }
      }
      __syncthreads();
    }
    if (tid == 0) {
      int i = ri[0];
      fidx[(size_t)b*P + it] = i;
      float x_ = px[i], y_ = py[i], z_ = pz[i];
      nxyz[((size_t)b*P+it)*3+0] = x_;
      nxyz[((size_t)b*P+it)*3+1] = y_;
      nxyz[((size_t)b*P+it)*3+2] = z_;
      cp[0] = x_; cp[1] = y_; cp[2] = z_;
    }
    __syncthreads();
    float cx = cp[0], cy = cp[1], cz = cp[2];
    for (int j = tid; j < N; j += 256) {
      float dx = px[j]-cx, dy = py[j]-cy, dz = pz[j]-cz;
      dist[j] = fminf(dist[j], dx*dx + dy*dy + dz*dz);
    }
    __syncthreads();
  }
}

// ---------------------------------------------------------------------------
// Ball query: one wave32 per center; ballot/popcount stream compaction keeps
// the first S in-radius indices in ascending order (== reference semantics).
// ---------------------------------------------------------------------------
__global__ void k_ballq(const float* __restrict__ xyz, const float* __restrict__ nxyz,
                        int N, int P, int S, float r2, int* __restrict__ gidx) {
  int cid = blockIdx.x;
  int b = cid / P, p = cid % P;
  int lane = threadIdx.x;
  const float* pb = xyz + (size_t)b * N * 3;
  const float* c  = nxyz + ((size_t)b * P + p) * 3;
  float cx = c[0], cy = c[1], cz = c[2];
  int* gdst = gidx + ((size_t)b * P + p) * S;
  int cnt = 0, first = N - 1;
  for (int base = 0; base < N; base += 32) {
    int j = base + lane;
    bool in = false;
    if (j < N) {
      float dx = pb[j*3+0]-cx, dy = pb[j*3+1]-cy, dz = pb[j*3+2]-cz;
      in = (dx*dx + dy*dy + dz*dz) < r2;
    }
    unsigned int m32 = (unsigned int)__ballot(in);
    if (in) {
      int slot = cnt + __popc(m32 & ((1u << lane) - 1u));
      if (slot < S) gdst[slot] = j;
    }
    if (cnt == 0 && m32 != 0u) first = base + (__ffs((int)m32) - 1);
    cnt += __popc(m32);
    if (cnt >= S) break;
  }
  int filled = cnt < S ? cnt : S;
  int fv = (cnt == 0) ? (N - 1) : first;
  for (int s = filled + lane; s < S; s += 32) gdst[s] = fv;
}

// ---------------------------------------------------------------------------
// Grouping kernels: build f16 (Kp, P*S) GEMM inputs (pad rows pre-zeroed).
// ---------------------------------------------------------------------------
__global__ void k_group1(const float* __restrict__ xyz, const float* __restrict__ nxyz,
                         const int* __restrict__ gidx, int N, int P, int S, int Kp,
                         half_t* __restrict__ X) {
  int cid = blockIdx.x;
  int b = cid / P, p = cid % P;
  int s = threadIdx.x;
  int idx = gidx[((size_t)b*P + p)*S + s];
  const float* pt = xyz  + ((size_t)b*N + idx)*3;
  const float* ce = nxyz + ((size_t)b*P + p)*3;
  size_t NS = (size_t)P * S;
  size_t n = (size_t)p * S + s;
  for (int c = 0; c < 3; ++c)
    X[((size_t)b*Kp + c)*NS + n] = (half_t)(pt[c] - ce[c]);
}

__global__ void k_group2(const float* __restrict__ xyz, const float* __restrict__ nxyz,
                         const int* __restrict__ gidx, const float* __restrict__ feat,
                         int N, int P, int S, int Cf, int Nprev, int Kp,
                         half_t* __restrict__ X) {
  int cid = blockIdx.x;
  int b = cid / P, p = cid % P;
  int s = threadIdx.x;
  int idx = gidx[((size_t)b*P + p)*S + s];
  const float* pt = xyz  + ((size_t)b*N + idx)*3;
  const float* ce = nxyz + ((size_t)b*P + p)*3;
  size_t NS = (size_t)P * S;
  size_t n = (size_t)p * S + s;
  for (int c = 0; c < 3; ++c)
    X[((size_t)b*Kp + c)*NS + n] = (half_t)(pt[c] - ce[c]);
  for (int cf = 0; cf < Cf; ++cf)
    X[((size_t)b*Kp + 3 + cf)*NS + n] =
        (half_t)feat[((size_t)b*Cf + cf)*Nprev + idx];
}

__global__ void k_group3(const float* __restrict__ xyz, const float* __restrict__ feat,
                         int Npts, int Cf, int Kp, half_t* __restrict__ X) {
  int b = blockIdx.x;
  int s = threadIdx.x;      // blockDim.x == Npts (128)
  for (int c = 0; c < 3; ++c)
    X[((size_t)b*Kp + c)*Npts + s] = (half_t)xyz[((size_t)b*Npts + s)*3 + c];
  for (int cf = 0; cf < Cf; ++cf)
    X[((size_t)b*Kp + 3 + cf)*Npts + s] = (half_t)feat[((size_t)b*Cf + cf)*Npts + s];
}

// ---------------------------------------------------------------------------
// WMMA GEMM + BN + ReLU (+ fused max-pool epilogue).
//   Out = relu( BN( Wh(MxKp,f16) * X(KpxN,f16) ) ), per batch (blockIdx.z).
// 256 threads = 8 waves; block tile 64(M) x 64(N); Kp stepped by 32; no edge
// guards (weights and activations pre-padded to Kp). Each wave runs two
// v_wmma_f32_16x16x32_f16 per K-step (two N-subtiles sharing the A fragment).
// ---------------------------------------------------------------------------
__global__ __launch_bounds__(256) void k_gemm(
    const half_t* __restrict__ Wh, const half_t* __restrict__ X,
    const float* __restrict__ gbn, const float* __restrict__ bbn,
    const float* __restrict__ mbn, const float* __restrict__ vbn,
    int M, int Kp, int N, half_t* __restrict__ Out,
    float* __restrict__ Pool, int S, int Pn, int mode) {
  __shared__ half_t As[64 * 32];      // [m][k], row-major
  __shared__ half_t Bs[64 * 34];      // transposed [n][k], stride 34
  int tid = threadIdx.x;
  int bb = blockIdx.z;
  int m0 = blockIdx.y * 64;
  int n0 = blockIdx.x * 64;
  const half_t* Xb = X + (size_t)bb * Kp * N;
  int lane = tid & 31, w = tid >> 5;
  int wm = w >> 1, wn = w & 1;
  int hi = lane >> 4, l15 = lane & 15;
  int am = wm * 16 + l15;
  int arow = tid >> 2, acb = (tid & 3) * 8;   // A stage: 64 rows x 32 cols
  int bk = tid >> 3,  nb  = (tid & 7) * 8;    // B stage: 32 k-rows x 64 n-cols
  v8f acc0 = {}, acc1 = {};
  for (int k0 = 0; k0 < Kp; k0 += 32) {
    // stage A: one 16B load + one ds_store_b128 per thread (no guards)
    *(v8h*)&As[arow * 32 + acb] =
        *(const v8h*)&Wh[(size_t)(m0 + arow) * Kp + k0 + acb];
    // stage B: one 16B global load, transpose-scatter into LDS
    v8h bx = *(const v8h*)(Xb + (size_t)(k0 + bk) * N + n0 + nb);
    #pragma unroll
    for (int i = 0; i < 8; ++i) Bs[(nb + i) * 34 + bk] = bx[i];
    __syncthreads();
    if (k0 + 32 < Kp)
      __builtin_prefetch(Xb + (size_t)(k0 + 32 + bk) * N + n0, 0, 1);
    // repack 16x32 (A) / 32x16 (B) with contiguous k = 16*hi + s in both
    v16h av, bv0, bv1;
    #pragma unroll
    for (int j = 0; j < 8; ++j) {
      int kb = 2 * j + hi * 16;
      av[2*j]   = As[am * 32 + kb];
      av[2*j+1] = As[am * 32 + kb + 1];
      int kbB = 2 * j + hi * 16;
      int r0 = (wn * 32 + l15) * 34 + kbB;
      int r1 = (wn * 32 + 16 + l15) * 34 + kbB;
      bv0[2*j]   = Bs[r0];
      bv0[2*j+1] = Bs[r0 + 1];
      bv1[2*j]   = Bs[r1];
      bv1[2*j+1] = Bs[r1 + 1];
    }
    acc0 = __builtin_amdgcn_wmma_f32_16x16x32_f16(false, av, false, bv0,
                                                  (short)0, acc0, false, false);
    acc1 = __builtin_amdgcn_wmma_f32_16x16x32_f16(false, av, false, bv1,
                                                  (short)0, acc1, false, false);
    __syncthreads();
  }
  // epilogue: BN + ReLU, then store f16 or fused max-pool (atomic)
  #pragma unroll
  for (int r = 0; r < 8; ++r) {
    int m = m0 + wm * 16 + r + hi * 8;
    float sc = gbn[m] * rsqrtf(vbn[m] + BN_EPS);
    float sh = bbn[m] - mbn[m] * sc;
    #pragma unroll
    for (int sub = 0; sub < 2; ++sub) {
      int n = n0 + wn * 32 + sub * 16 + l15;
      float y = fmaxf(fmaf(sub ? acc1[r] : acc0[r], sc, sh), 0.0f);
      if (mode == MODE_STORE) {
        Out[(size_t)bb * M * N + (size_t)m * N + n] = (half_t)y;
      } else {
        // 16 lanes of a half-wave share (m, n/S): reduce, then one atomic
        #pragma unroll
        for (int off = 1; off < 16; off <<= 1) y = fmaxf(y, __shfl_xor(y, off, 32));
        if (l15 == 0) {
          int p = n / S;
          atomicMax((int*)(Pool + ((size_t)bb * M + m) * Pn + p), __float_as_int(y));
        }
      }
    }
  }
}

// ---------------------------------------------------------------------------
// FC head (tiny): one thread per (batch, out-channel).
// mode 0: BN + ReLU; mode 1: +bias (bias passed via `g`), no activation.
// ---------------------------------------------------------------------------
__global__ void k_fc(const float* __restrict__ in, const float* __restrict__ W,
                     const float* __restrict__ g, const float* __restrict__ bbn,
                     const float* __restrict__ mbn, const float* __restrict__ vbn,
                     int B, int M, int K, float* __restrict__ out, int mode) {
  int idx = blockIdx.x * blockDim.x + threadIdx.x;
  if (idx >= B * M) return;
  int b = idx / M, o = idx % M;
  const float* xi = in + (size_t)b * K;
  const float* wo = W + (size_t)o * K;
  float acc = 0.0f;
  for (int k = 0; k < K; ++k) acc = fmaf(xi[k], wo[k], acc);
  if (mode == 0) {
    float sc = g[o] * rsqrtf(vbn[o] + BN_EPS);
    acc = fmaxf(fmaf(acc, sc, bbn[o] - mbn[o] * sc), 0.0f);
  } else {
    acc += g[o];
  }
  out[idx] = acc;
}

// log_softmax per row; one wave per batch row, Kc <= 64.
__global__ void k_logsm(const float* __restrict__ logits, float* __restrict__ out, int Kc) {
  int b = blockIdx.x, lane = threadIdx.x;
  const float* r = logits + (size_t)b * Kc;
  float v0 = (lane      < Kc) ? r[lane]      : -1e30f;
  float v1 = (lane + 32 < Kc) ? r[lane + 32] : -1e30f;
  float mx = fmaxf(v0, v1);
  #pragma unroll
  for (int off = 1; off < 32; off <<= 1) mx = fmaxf(mx, __shfl_xor(mx, off, 32));
  float s = 0.0f;
  if (lane      < Kc) s += expf(v0 - mx);
  if (lane + 32 < Kc) s += expf(v1 - mx);
  #pragma unroll
  for (int off = 1; off < 32; off <<= 1) s += __shfl_xor(s, off, 32);
  float lse = mx + logf(s);
  if (lane      < Kc) out[(size_t)b * Kc + lane]      = v0 - lse;
  if (lane + 32 < Kc) out[(size_t)b * Kc + lane + 32] = v1 - lse;
}

// ---------------------------------------------------------------------------
extern "C" void kernel_launch(void* const* d_in, const int* in_sizes, int n_in,
                              void* d_out, int out_size, void* d_ws, size_t ws_size,
                              hipStream_t stream) {
  (void)in_sizes; (void)n_in; (void)out_size; (void)ws_size;
  const int B = 16, N1 = 4096, P1 = 512, P2 = 128, S = 64, KCLS = 40;
  // padded K per layer (multiples of 32)
  const int KP10 = 32, KP11 = 64, KP20 = 160, KP21 = 128, KP30 = 288, KP31 = 512;

  const float* x  = (const float*)d_in[0];
  const float* x2 = (const float*)d_in[1];
  const float *W10=(const float*)d_in[2],  *g10=(const float*)d_in[3],  *b10=(const float*)d_in[4],  *m10=(const float*)d_in[5],  *v10=(const float*)d_in[6];
  const float *W11=(const float*)d_in[7],  *g11=(const float*)d_in[8],  *b11=(const float*)d_in[9],  *m11=(const float*)d_in[10], *v11=(const float*)d_in[11];
  const float *W20=(const float*)d_in[12], *g20=(const float*)d_in[13], *b20=(const float*)d_in[14], *m20=(const float*)d_in[15], *v20=(const float*)d_in[16];
  const float *W21=(const float*)d_in[17], *g21=(const float*)d_in[18], *b21=(const float*)d_in[19], *m21=(const float*)d_in[20], *v21=(const float*)d_in[21];
  const float *W30=(const float*)d_in[22], *g30=(const float*)d_in[23], *b30=(const float*)d_in[24], *m30=(const float*)d_in[25], *v30=(const float*)d_in[26];
  const float *W31=(const float*)d_in[27], *g31=(const float*)d_in[28], *b31=(const float*)d_in[29], *m31=(const float*)d_in[30], *v31=(const float*)d_in[31];
  const float *FW1=(const float*)d_in[32];
  const float *fg1=(const float*)d_in[33], *fb1=(const float*)d_in[34], *fm1=(const float*)d_in[35], *fv1=(const float*)d_in[36];
  const float *FW2=(const float*)d_in[37];
  const float *fg2=(const float*)d_in[38], *fb2=(const float*)d_in[39], *fm2=(const float*)d_in[40], *fv2=(const float*)d_in[41];
  const float *FW3=(const float*)d_in[42], *fb3=(const float*)d_in[43];

  // d_out: [logits 16*40][globF 16*1024][globF2 16*1024]
  float* out    = (float*)d_out;
  float* globF  = out + B * KCLS;
  float* globF2 = globF + B * 1024;

  char* wsp = (char*)d_ws;
  size_t off = 0;
  auto alloc = [&](size_t bytes) -> void* {
    void* p = wsp + off;
    off = (off + bytes + 255) & ~(size_t)255;
    return p;
  };
  int*    fidx1   = (int*)   alloc((size_t)B * P1 * 4);
  float*  nxyz1   = (float*) alloc((size_t)B * P1 * 3 * 4);
  int*    gidx1   = (int*)   alloc((size_t)B * P1 * S * 4);
  int*    fidx2   = (int*)   alloc((size_t)B * P2 * 4);
  float*  nxyz2   = (float*) alloc((size_t)B * P2 * 3 * 4);
  int*    gidx2   = (int*)   alloc((size_t)B * P2 * S * 4);
  float*  pooled1 = (float*) alloc((size_t)B * 128 * P1 * 4);
  float*  pooled2 = (float*) alloc((size_t)B * 256 * P2 * 4);
  half_t* Wh10    = (half_t*)alloc((size_t)64   * KP10 * 2);
  half_t* Wh11    = (half_t*)alloc((size_t)128  * KP11 * 2);
  half_t* Wh20    = (half_t*)alloc((size_t)128  * KP20 * 2);
  half_t* Wh21    = (half_t*)alloc((size_t)256  * KP21 * 2);
  half_t* Wh30    = (half_t*)alloc((size_t)512  * KP30 * 2);
  half_t* Wh31    = (half_t*)alloc((size_t)1024 * KP31 * 2);
  size_t  szX1 = (size_t)B * KP10 * (P1 * S) * 2;   // 33.5 MB
  size_t  szX2 = (size_t)B * KP20 * (P2 * S) * 2;   // 42 MB
  size_t  szX3 = (size_t)B * KP30 * P2 * 2;
  size_t  szXmax = szX2 > szX1 ? szX2 : szX1;
  half_t* Xbuf    = (half_t*)alloc(szXmax);
  half_t* Abuf    = (half_t*)alloc((size_t)B * 64 * (P1 * S) * 2);  // 67 MB (max act)
  float*  h1      = (float*) alloc((size_t)B * 512 * 4);
  float*  h2      = (float*) alloc((size_t)B * 256 * 4);
  float*  lg      = (float*) alloc((size_t)B * KCLS * 4);

  // one-time (per call) weight conversion to padded f16
  k_cvtw<<<(64*KP10   + 255)/256, 256, 0, stream>>>(W10, Wh10, 64,   3,   KP10);
  k_cvtw<<<(128*KP11  + 255)/256, 256, 0, stream>>>(W11, Wh11, 128,  64,  KP11);
  k_cvtw<<<(128*KP20  + 255)/256, 256, 0, stream>>>(W20, Wh20, 128,  131, KP20);
  k_cvtw<<<(256*KP21  + 255)/256, 256, 0, stream>>>(W21, Wh21, 256,  128, KP21);
  k_cvtw<<<(512*KP30  + 255)/256, 256, 0, stream>>>(W30, Wh30, 512,  259, KP30);
  k_cvtw<<<(1024*KP31 + 255)/256, 256, 0, stream>>>(W31, Wh31, 1024, 512, KP31);

  size_t fps_sh1 = ((size_t)4 * N1 + 256) * 4 + 256 * 4 + 16;
  size_t fps_sh2 = ((size_t)4 * P1 + 256) * 4 + 256 * 4 + 16;

  auto backbone = [&](const float* xin, float* gF) {
    int N1s = P1 * S;   // 32768
    int N2s = P2 * S;   // 8192
    // ---- SA1 ----
    k_fps  <<<B, 256, fps_sh1, stream>>>(xin, N1, P1, fidx1, nxyz1);
    k_ballq<<<B * P1, 32, 0, stream>>>(xin, nxyz1, N1, P1, S, 0.04f, gidx1);
    hipMemsetAsync(Xbuf, 0, szX1, stream);
    k_group1<<<B * P1, S, 0, stream>>>(xin, nxyz1, gidx1, N1, P1, S, KP10, Xbuf);
    k_gemm<<<dim3(N1s/64, 64/64, B), 256, 0, stream>>>(
        Wh10, Xbuf, g10, b10, m10, v10, 64, KP10, N1s, Abuf, nullptr, 1, 1, MODE_STORE);
    hipMemsetAsync(pooled1, 0, (size_t)B * 128 * P1 * 4, stream);
    k_gemm<<<dim3(N1s/64, 128/64, B), 256, 0, stream>>>(
        Wh11, Abuf, g11, b11, m11, v11, 128, KP11, N1s, nullptr, pooled1, S, P1, MODE_POOL);
    // ---- SA2 ----
    k_fps  <<<B, 256, fps_sh2, stream>>>(nxyz1, P1, P2, fidx2, nxyz2);
    k_ballq<<<B * P2, 32, 0, stream>>>(nxyz1, nxyz2, P1, P2, S, 0.16f, gidx2);
    hipMemsetAsync(Xbuf, 0, szX2, stream);
    k_group2<<<B * P2, S, 0, stream>>>(nxyz1, nxyz2, gidx2, pooled1, P1, P2, S, 128, P1, KP20, Xbuf);
    k_gemm<<<dim3(N2s/64, 128/64, B), 256, 0, stream>>>(
        Wh20, Xbuf, g20, b20, m20, v20, 128, KP20, N2s, Abuf, nullptr, 1, 1, MODE_STORE);
    hipMemsetAsync(pooled2, 0, (size_t)B * 256 * P2 * 4, stream);
    k_gemm<<<dim3(N2s/64, 256/64, B), 256, 0, stream>>>(
        Wh21, Abuf, g21, b21, m21, v21, 256, KP21, N2s, nullptr, pooled2, S, P2, MODE_POOL);
    // ---- SA3 (group all 128 points, P=1, S=128) ----
    hipMemsetAsync(Xbuf, 0, szX3, stream);
    k_group3<<<B, P2, 0, stream>>>(nxyz2, pooled2, P2, 256, KP30, Xbuf);
    k_gemm<<<dim3(P2/64, 512/64, B), 256, 0, stream>>>(
        Wh30, Xbuf, g30, b30, m30, v30, 512, KP30, P2, Abuf, nullptr, 1, 1, MODE_STORE);
    hipMemsetAsync(gF, 0, (size_t)B * 1024 * 4, stream);
    k_gemm<<<dim3(P2/64, 1024/64, B), 256, 0, stream>>>(
        Wh31, Abuf, g31, b31, m31, v31, 1024, KP31, P2, nullptr, gF, P2, 1, MODE_POOL);
  };

  backbone(x,  globF);
  backbone(x2, globF2);

  // FC head (uses globF only, per reference)
  k_fc<<<(B*512 + 255)/256, 256, 0, stream>>>(globF, FW1, fg1, fb1, fm1, fv1, B, 512, 1024, h1, 0);
  k_fc<<<(B*256 + 255)/256, 256, 0, stream>>>(h1,    FW2, fg2, fb2, fm2, fv2, B, 256, 512,  h2, 0);
  k_fc<<<(B*KCLS + 255)/256, 256, 0, stream>>>(h2,   FW3, fb3, nullptr, nullptr, nullptr, B, KCLS, 256, lg, 1);
  k_logsm<<<B, 32, 0, stream>>>(lg, out, KCLS);
}
